// QLSTM_65481071403358
// MI455X (gfx1250) — matrix operand order repaired and checked by reading, not compiled, not run on hardware
//
#include <hip/hip_runtime.h>
#include <hip/hip_bf16.h>
#include <cmath>

#define NQ        8
#define QDIM      256
#define TSTEPS    256
#define BATCH     1024
#define D_IN      264
#define NGATES    4

typedef __attribute__((ext_vector_type(16))) _Float16 v16h;
typedef __attribute__((ext_vector_type(8)))  _Float16 v8h;
typedef __attribute__((ext_vector_type(8)))  float    v8f;

static __device__ __forceinline__ v16h cat16(v8h a, v8h b) {
  return __builtin_shufflevector(a, b, 0,1,2,3,4,5,6,7,8,9,10,11,12,13,14,15);
}
static __device__ __forceinline__ v8f wmma_f16(v16h a, v16h b, v8f c) {
  // D = A(16x32 f16) x B(32x16 f16) + C(16x16 f32)
  return __builtin_amdgcn_wmma_f32_16x16x32_f16(false, a, false, b, (short)0, c, false, false);
}
static __device__ __forceinline__ v8h cvt8(float4 a, float4 b) {
  v8h r;
  r[0]=(_Float16)a.x; r[1]=(_Float16)a.y; r[2]=(_Float16)a.z; r[3]=(_Float16)a.w;
  r[4]=(_Float16)b.x; r[5]=(_Float16)b.y; r[6]=(_Float16)b.z; r[7]=(_Float16)b.w;
  return r;
}

// L-nibbles (wires 4..7) with even / odd popcount, ascending: compact ranks are
// consecutive within each 16-index block, so each run packs to one ds_store_b128.
static constexpr int EVL[8] = {0,3,5,6,9,10,12,15};
static constexpr int ODL[8] = {1,2,4,7,8,11,13,14};

// Build psi0 fragments (A-operand layout) for one gate. Lane handles sample s,
// H-nibbles [HB, HB+8). psi layout: [part(2)][ktile(4)][lane(32)][16 halves].
template<int HB>
__device__ __forceinline__ void buildPsiHalf(const float (&p01)[4], const float (&p23)[4],
                                             const float (&pl)[16], _Float16* psiW, int s) {
#pragma unroll
  for (int hh = 0; hh < 8; ++hh) {
    const int H   = HB + hh;
    const int pcH = __builtin_popcount(H);
    const float ph = p01[H >> 2] * p23[H & 3];
    v8h r0, r1;
#pragma unroll
    for (int i = 0; i < 8; ++i) {
      const int L0 = (pcH & 1) ? ODL[i] : EVL[i];   // total popcount even -> real part
      const int L1 = (pcH & 1) ? EVL[i] : ODL[i];   // total popcount odd  -> imag part
      const float a0 = ph * pl[L0];
      const float a1 = ph * pl[L1];
      const int q0 = (pcH + __builtin_popcount(L0)) & 3;  // 0 or 2 : (-i)^pc = +/-1
      const int q1 = (pcH + __builtin_popcount(L1)) & 3;  // 1 or 3 : (-i)^pc = -/+i
      r0[i] = (_Float16)((q0 == 0) ?  a0 : -a0);
      r1[i] = (_Float16)((q1 == 3) ?  a1 : -a1);
    }
    const int kt    = H >> 2;
    const int lane2 = s + ((H & 2) ? 16 : 0);
    const int io    = (H & 1) ? 8 : 0;
    *(v8h*)(psiW + (((0*4 + kt)*32 + lane2) << 4) + io) = r0;
    *(v8h*)(psiW + (((1*4 + kt)*32 + lane2) << 4) + io) = r1;
  }
}

// ---------------------------------------------------------------------------
// Kernel 0: pack the input-projection weights once so the hot kernels never
// touch runtime-selected (flat) pointers:
//   WxPack : 16 B-fragments (nt 0..1, kt 0..7) of Wx^T [256 x 32] in f16,
//            fragment-linear [frag][lane(32)][16 halves]
//   WhPack : [32][8] f32 recurrent slices Wg[w][256..263]
//   BPack  : [32]    f32 biases
// ---------------------------------------------------------------------------
__global__ void qlstm_prepW(const float* __restrict__ Wf, const float* __restrict__ bf,
                            const float* __restrict__ Wi, const float* __restrict__ bi,
                            const float* __restrict__ Wu, const float* __restrict__ bu,
                            const float* __restrict__ Wo, const float* __restrict__ bo,
                            _Float16* __restrict__ WxPack, float* __restrict__ WhPack,
                            float* __restrict__ BPack) {
  const int tid = threadIdx.x;
  const float* Ws[4] = {Wf, Wi, Wu, Wo};
  const float* Bs[4] = {bf, bi, bu, bo};
  const int vl = tid & 31;
  for (int fi = tid >> 5; fi < 16; fi += 8) {
    const int nt = fi & 1, kt = fi >> 1;
    const int n  = nt*16 + (vl & 15);
    const float* wrow = Ws[n >> 3] + (n & 7)*D_IN;
    const int kb = kt*32 + ((vl < 16) ? 0 : 8);
    v8h r0, r1;
#pragma unroll
    for (int i = 0; i < 8; ++i) {
      r0[i] = (_Float16)wrow[kb + i];
      r1[i] = (_Float16)wrow[kb + 16 + i];
    }
    *(v8h*)(WxPack + fi*512 + vl*16)     = r0;
    *(v8h*)(WxPack + fi*512 + vl*16 + 8) = r1;
  }
  {
    const int n = tid >> 3, j = tid & 7;         // tid 0..255
    WhPack[tid] = Ws[n >> 3][(n & 7)*D_IN + 256 + j];
  }
  if (tid < 32) BPack[tid] = Bs[tid >> 3][tid & 7];
}

// ---------------------------------------------------------------------------
// Kernel 1: build the fixed per-gate 256x256 real unitary U (DEPTH x (RY + CNOT
// ring)) by simulating basis columns in LDS, then pack WMMA B-fragments of
// U^T restricted to even/odd popcount columns (compact K=128), f16.
// One block per gate; thread j owns statevector column j (no barriers needed
// during the circuit since 1q gates and CNOTs act within a column).
// ---------------------------------------------------------------------------
__global__ void qlstm_prepU(const float* __restrict__ pf, const float* __restrict__ pi_,
                            const float* __restrict__ pu, const float* __restrict__ po,
                            _Float16* __restrict__ Upack) {
  extern __shared__ char smem[];
  float* S = (float*)smem;                                      // 256*256 f32
  unsigned short* perm = (unsigned short*)(smem + QDIM*QDIM*4); // [2][128]
  const int tid = threadIdx.x;
  const int g   = blockIdx.x;
  const float* pg = (g == 0) ? pf : (g == 1) ? pi_ : (g == 2) ? pu : po;

  for (int idx = tid; idx < QDIM*QDIM; idx += 256)
    S[idx] = ((idx >> 8) == (idx & 255)) ? 1.f : 0.f;
  if (tid == 0) {
    int e = 0, o = 0;
    for (int j = 0; j < QDIM; ++j) {
      if (__builtin_popcount(j) & 1) perm[128 + (o++)] = (unsigned short)j;
      else                           perm[e++]         = (unsigned short)j;
    }
  }
  __syncthreads();

  {
    const int j = tid;
    for (int d = 0; d < 2; ++d) {
      for (int w = 0; w < NQ; ++w) {                // RY(params[d*8+w]) on wire w
        const float th = pg[d*NQ + w] * 0.5f;
        float sth, cth; __sincosf(th, &sth, &cth);
        const int bit = 1 << (7 - w);
        for (int base = 0; base < 128; ++base) {
          const int k0 = ((base & ~(bit - 1)) << 1) | (base & (bit - 1));
          const int k1 = k0 | bit;
          const float a0 = S[k0*QDIM + j], a1 = S[k1*QDIM + j];
          S[k0*QDIM + j] = cth*a0 - sth*a1;
          S[k1*QDIM + j] = sth*a0 + cth*a1;
        }
      }
      for (int w = 0; w < NQ; ++w) {                // CNOT ring (7,0 last)
        const int cbit = 1 << (7 - w);
        const int tbit = 1 << (7 - ((w + 1) & 7));
        for (int k = 0; k < QDIM; ++k) {
          if ((k & cbit) && !(k & tbit)) {
            const float tmp = S[k*QDIM + j];
            S[k*QDIM + j] = S[(k | tbit)*QDIM + j];
            S[(k | tbit)*QDIM + j] = tmp;
          }
        }
      }
    }
  }
  __syncthreads();

  // Pack B fragments: B[kc][n] = U[n][perm[p][kc]]; frag = [lane(32)][16 halves]
  const int vl = tid & 31;
  for (int f2 = tid >> 5; f2 < 128; f2 += 8) {
    const int p  = f2 >> 6;
    const int kt = (f2 >> 4) & 3;
    const int nt = f2 & 15;
    const int n  = nt*16 + (vl & 15);
    const int kb = kt*32 + ((vl < 16) ? 0 : 16);
    v8h r0, r1;
#pragma unroll
    for (int i = 0; i < 8; ++i) {
      r0[i] = (_Float16)S[n*QDIM + perm[p*128 + kb + i]];
      r1[i] = (_Float16)S[n*QDIM + perm[p*128 + kb + 8 + i]];
    }
    const size_t off = ((size_t)((g*2 + p)*4 + kt)*16 + nt)*512 + (size_t)vl*16;
    *(v8h*)(Upack + off)     = r0;
    *(v8h*)(Upack + off + 8) = r1;
  }
}

// ---------------------------------------------------------------------------
// Kernel 2: xang[t*B+s, g*8+w] = x[t,s,:] . Wg[w,0:256] + bg[w]  (f16 WMMA GEMM)
// One wave per 16-row tile; N = 32 (4 gates x 8 wires), K = 256.
// B-operands come pre-packed from WxPack (contiguous global_load_b128 pairs).
// ---------------------------------------------------------------------------
__global__ void qlstm_prepX(const float* __restrict__ X,
                            const _Float16* __restrict__ WxPack,
                            const float* __restrict__ BPack,
                            float* __restrict__ xang) {
  const int tid = threadIdx.x;
  const int wv = tid >> 5, l = tid & 31, ln = l & 15, lh = l >> 4;
  const long tile = (long)blockIdx.x*8 + wv;
  const long row0 = tile * 16;

  v16h bw[2][8];
#pragma unroll
  for (int nt = 0; nt < 2; ++nt)
#pragma unroll
    for (int kt = 0; kt < 8; ++kt) {
      const _Float16* bp = WxPack + (kt*2 + nt)*512 + l*16;
      bw[nt][kt] = cat16(*(const v8h*)bp, *(const v8h*)(bp + 8));
    }

  v8f acc0 = {}, acc1 = {};
  const float* arow = X + (size_t)(row0 + ln)*256;
#pragma unroll
  for (int kt = 0; kt < 8; ++kt) {
    const int kb = kt*32 + (lh ? 8 : 0);
    float4 a0 = *(const float4*)(arow + kb);
    float4 a1 = *(const float4*)(arow + kb + 4);
    float4 a2 = *(const float4*)(arow + kb + 16);
    float4 a3 = *(const float4*)(arow + kb + 20);
    v16h af = cat16(cvt8(a0, a1), cvt8(a2, a3));
    acc0 = wmma_f16(af, bw[0][kt], acc0);
    acc1 = wmma_f16(af, bw[1][kt], acc1);
  }
  v8f accs[2] = {acc0, acc1};
#pragma unroll
  for (int nt = 0; nt < 2; ++nt) {
    const int n = nt*16 + ln;
    const float bv = BPack[n];
#pragma unroll
    for (int r = 0; r < 8; ++r) {
      const long row = row0 + r + (lh ? 8 : 0);
      xang[row*32 + n] = accs[nt][r] + bv;
    }
  }
}

// ---------------------------------------------------------------------------
// Kernel 3: persistent recurrence. 16 blocks x 128 threads (4 waves);
// each wave owns 16 batch samples for all T steps (wave-private LDS, no
// barriers in the loop: per-wave DS ops are in-order on CDNA5).
// Per step per gate: build psi0 fragments; a = psi_re @ U_E^T, b = psi_im @
// U_O^T via f16 WMMA (K=128, N=256, streamed from L2-resident packed U);
// probs = a^2+b^2 staged to LDS in A-layout f16; z = probs @ SIGNS via WMMA.
// ---------------------------------------------------------------------------
__global__ void qlstm_main(const float* __restrict__ xang,
                           const _Float16* __restrict__ Upack,
                           const float* __restrict__ WhPack,
                           float* __restrict__ out) {
  extern __shared__ char smem[];
  const int tid = threadIdx.x;
  const int wv  = tid >> 5;
  const int l   = tid & 31;
  const int ln  = l & 15;
  const int lh  = l >> 4;
  const int stile = blockIdx.x*4 + wv;          // 0..63 -> samples [stile*16, +16)

  const int WSZ = 21504;
  char* wb = smem + wv*WSZ;
  _Float16* psiW   = (_Float16*)(wb);           // 8 KB  [2][4][32][16] halves
  _Float16* probsW = (_Float16*)(wb + 8192);    // 8 KB  [8][32][16] halves (A layout)
  float* angW = (float*)(wb + 16384);           // 2 KB  [16][32]
  float* zW   = (float*)(wb + 18432);           // 2 KB  [4][16][8]
  float* hW   = (float*)(wb + 20480);           // 512 B [16][8]
  float* cW   = (float*)(wb + 20992);           // 512 B [16][8]
  float* WhLds = (float*)(smem + 4*WSZ);        // 1 KB shared: [32][8] h-weights

#pragma unroll
  for (int e = 0; e < 2; ++e) {                 // single-base pointer -> global_load
    const int idx = tid*2 + e;                  // 0..255
    WhLds[idx] = WhPack[idx];
  }
#pragma unroll
  for (int e = 0; e < 4; ++e) { hW[l*4 + e] = 0.f; cW[l*4 + e] = 0.f; }
  __syncthreads();

  // SIGNS B-fragments (K=256, N: cols 0..7 = wires, 8..15 zero-padded)
  v16h sb[8];
#pragma unroll
  for (int kt2 = 0; kt2 < 8; ++kt2) {
    v8h x0, x1;
#pragma unroll
    for (int i = 0; i < 8; ++i) {
      const int k0 = kt2*32 + (lh ? 8 : 0) + i;
      const int k1 = k0 + 16;
      const float s0 = (ln < 8) ? (((k0 >> (7 - ln)) & 1) ? -1.f : 1.f) : 0.f;
      const float s1 = (ln < 8) ? (((k1 >> (7 - ln)) & 1) ? -1.f : 1.f) : 0.f;
      x0[i] = (_Float16)s0; x1[i] = (_Float16)s1;
    }
    sb[kt2] = cat16(x0, x1);
  }

  const size_t TB8 = (size_t)TSTEPS*BATCH*8;

#pragma unroll 1
  for (int t = 0; t < TSTEPS; ++t) {
    // ---- angles: xang + h @ Wh^T ----
    {
      const int s = l >> 1, n0 = (l & 1)*16;
      const float* xa = xang + (((size_t)t*BATCH + (size_t)stile*16 + s)*32 + n0);
      if (t + 1 < TSTEPS)                         // warm L2/WGP$ for next step
        __builtin_prefetch(xa + (size_t)BATCH*32, 0, 1);
      float xs[16];
      *(float4*)(xs)      = ((const float4*)xa)[0];
      *(float4*)(xs + 4)  = ((const float4*)xa)[1];
      *(float4*)(xs + 8)  = ((const float4*)xa)[2];
      *(float4*)(xs + 12) = ((const float4*)xa)[3];
      float4 ha = *(const float4*)&hW[s*8];
      float4 hb = *(const float4*)&hW[s*8 + 4];
#pragma unroll
      for (int q = 0; q < 16; ++q) {
        const int n = n0 + q;
        float4 wa = *(const float4*)&WhLds[n*8];
        float4 wbv = *(const float4*)&WhLds[n*8 + 4];
        float acc = xs[q] + ha.x*wa.x + ha.y*wa.y + ha.z*wa.z + ha.w*wa.w
                          + hb.x*wbv.x + hb.y*wbv.y + hb.z*wbv.z + hb.w*wbv.w;
        angW[s*32 + n] = acc;
      }
    }

#pragma unroll 1
    for (int g = 0; g < NGATES; ++g) {
      // ---- psi0 product state, fragment-packed into LDS ----
      {
        const int s = ln;
        float cc[NQ], ssn[NQ];
#pragma unroll
        for (int w = 0; w < NQ; ++w) {
          const float a = angW[s*32 + g*8 + w]*0.5f;
          __sincosf(a, &ssn[w], &cc[w]);
        }
        float p01[4] = {cc[0]*cc[1], cc[0]*ssn[1], ssn[0]*cc[1], ssn[0]*ssn[1]};
        float p23[4] = {cc[2]*cc[3], cc[2]*ssn[3], ssn[2]*cc[3], ssn[2]*ssn[3]};
        float p45[4] = {cc[4]*cc[5], cc[4]*ssn[5], ssn[4]*cc[5], ssn[4]*ssn[5]};
        float p67[4] = {cc[6]*cc[7], cc[6]*ssn[7], ssn[6]*cc[7], ssn[6]*ssn[7]};
        float pl[16];
#pragma unroll
        for (int L = 0; L < 16; ++L) pl[L] = p45[L >> 2]*p67[L & 3];
        if (lh == 0) buildPsiHalf<0>(p01, p23, pl, psiW, s);
        else         buildPsiHalf<8>(p01, p23, pl, psiW, s);
      }

      // ---- a/b GEMMs + probs staging ----
      v16h apsi[2][4];
#pragma unroll
      for (int p = 0; p < 2; ++p)
#pragma unroll
        for (int kt = 0; kt < 4; ++kt) {
          const _Float16* ap = psiW + (((p*4 + kt)*32 + l) << 4);
          apsi[p][kt] = cat16(*(const v8h*)ap, *(const v8h*)(ap + 8));
        }
      const _Float16* ug = Upack + (size_t)g*65536;

#pragma unroll 2
      for (int nt = 0; nt < 16; ++nt) {
        v8f aacc = {}, bacc = {};
#pragma unroll
        for (int kt = 0; kt < 4; ++kt) {
          const _Float16* b0 = ug + (size_t)(((0*4 + kt)*16 + nt)*512) + l*16;
          aacc = wmma_f16(apsi[0][kt], cat16(*(const v8h*)b0, *(const v8h*)(b0 + 8)), aacc);
          const _Float16* b1 = ug + (size_t)(((1*4 + kt)*16 + nt)*512) + l*16;
          bacc = wmma_f16(apsi[1][kt], cat16(*(const v8h*)b1, *(const v8h*)(b1 + 8)), bacc);
        }
#pragma unroll
        for (int r = 0; r < 8; ++r) {
          const float pv = aacc[r]*aacc[r] + bacc[r]*bacc[r];
          const int m   = r + (lh ? 8 : 0);
          const int col = nt*16 + ln;
          const int kk  = col & 31;
          const int kt2 = col >> 5;
          const int lane2 = m + ((((kk >= 8) && (kk < 16)) || (kk >= 24)) ? 16 : 0);
          const int i2 = (kk < 8) ? kk : (kk < 16) ? (kk - 8) : (kk < 24) ? (kk - 8) : (kk - 16);
          probsW[((kt2*32 + lane2) << 4) + i2] = (_Float16)pv;
        }
      }

      // ---- z = probs @ SIGNS via WMMA ----
      v8f zacc = {};
#pragma unroll
      for (int kt2 = 0; kt2 < 8; ++kt2) {
        const _Float16* pp = probsW + ((kt2*32 + l) << 4);
        zacc = wmma_f16(cat16(*(const v8h*)pp, *(const v8h*)(pp + 8)), sb[kt2], zacc);
      }
      if (ln < 8) {
#pragma unroll
        for (int r = 0; r < 8; ++r) {
          const int m = r + (lh ? 8 : 0);
          zW[(g*16 + m)*8 + ln] = zacc[r];
        }
      }
    }

    // ---- LSTM elementwise update ----
#pragma unroll
    for (int e = 0; e < 4; ++e) {
      const int p = l*4 + e;
      const int s = p >> 3, w = p & 7;
      const float zf = zW[(0*16 + s)*8 + w];
      const float zi = zW[(1*16 + s)*8 + w];
      const float zu = zW[(2*16 + s)*8 + w];
      const float zo = zW[(3*16 + s)*8 + w];
      const float fv = 1.f/(1.f + __expf(-zf));
      const float iv = 1.f/(1.f + __expf(-zi));
      const float gv = tanhf(zu);
      const float ov = 1.f/(1.f + __expf(-zo));
      const float cv = fv*cW[s*8 + w] + iv*gv;
      const float hv = ov*tanhf(cv);
      cW[s*8 + w] = cv;
      hW[s*8 + w] = hv;
      const size_t gS = (size_t)stile*16 + s;
      out[((size_t)t*BATCH + gS)*8 + w] = hv;
      if (t == TSTEPS - 1) {
        out[TB8 + gS*8 + w]                   = hv;   // hx
        out[TB8 + (size_t)BATCH*8 + gS*8 + w] = cv;   // cx
      }
    }
  }
}

extern "C" void kernel_launch(void* const* d_in, const int* in_sizes, int n_in,
                              void* d_out, int out_size, void* d_ws, size_t ws_size,
                              hipStream_t stream) {
  (void)in_sizes; (void)n_in; (void)out_size; (void)ws_size;
  const float* X  = (const float*)d_in[0];
  const float* Wf = (const float*)d_in[1];  const float* bf = (const float*)d_in[2];
  const float* Wi = (const float*)d_in[3];  const float* bi = (const float*)d_in[4];
  const float* Wu = (const float*)d_in[5];  const float* bu = (const float*)d_in[6];
  const float* Wo = (const float*)d_in[7];  const float* bo = (const float*)d_in[8];
  const float* pf = (const float*)d_in[9];  const float* pi_ = (const float*)d_in[10];
  const float* pu = (const float*)d_in[11]; const float* po  = (const float*)d_in[12];
  float* out = (float*)d_out;

  char* ws = (char*)d_ws;
  _Float16* Upack  = (_Float16*)ws;                        // 512 KB (L2-resident)
  _Float16* WxPack = (_Float16*)(ws + 512*1024);           // 16 KB
  float*    WhPack = (float*)(ws + 528*1024);              // 1 KB
  float*    BPack  = (float*)(ws + 529*1024);              // 128 B
  float*    xang   = (float*)(ws + 1024*1024);             // 32 MB

  const int smemU = QDIM*QDIM*4 + 512;     // 256 KB state + perm LUT
  const int smemM = 4*21504 + 1024;        // ~85 KB
  hipFuncSetAttribute(reinterpret_cast<const void*>(qlstm_prepU),
                      hipFuncAttributeMaxDynamicSharedMemorySize, smemU);
  hipFuncSetAttribute(reinterpret_cast<const void*>(qlstm_main),
                      hipFuncAttributeMaxDynamicSharedMemorySize, smemM);

  qlstm_prepW<<<1, 256, 0, stream>>>(Wf, bf, Wi, bi, Wu, bu, Wo, bo,
                                     WxPack, WhPack, BPack);
  qlstm_prepU<<<NGATES, 256, smemU, stream>>>(pf, pi_, pu, po, Upack);
  qlstm_prepX<<<2048, 256, 0, stream>>>(X, WxPack, BPack, xang);
  qlstm_main<<<16, 128, smemM, stream>>>(xang, Upack, WhPack, out);
}
